// GCN_60945585930926
// MI455X (gfx1250) — compile-verified, matrix operand order checked by reading, not run on hardware
//
#include <hip/hip_runtime.h>
#include <stdint.h>

#define DIM 128

typedef __attribute__((ext_vector_type(16))) _Float16 v16h;
typedef __attribute__((ext_vector_type(8)))  _Float16 v8h;
typedef __attribute__((ext_vector_type(8)))  float    v8f;

// ---------------------------------------------------------------- utilities

__global__ void k_zero_f4(float4* __restrict__ p, long n4) {
  long i = (long)blockIdx.x * blockDim.x + threadIdx.x;
  if (i < n4) p[i] = make_float4(0.f, 0.f, 0.f, 0.f);
}

__global__ void k_count(const int* __restrict__ dst, float* __restrict__ deg, int E) {
  int e = blockIdx.x * blockDim.x + threadIdx.x;
  if (e < E) atomicAdd(deg + dst[e], 1.0f);
}

__global__ void k_inv(float* __restrict__ deg, int n) {
  int i = blockIdx.x * blockDim.x + threadIdx.x;
  if (i < n) deg[i] = 1.0f / fmaxf(deg[i], 1.0f);
}

// Fuse Wl,Wr (each [128,128] f32, row-major, row = output channel) into
// Wf[r][0..127] = Wl row r, Wf[r][128..255] = Wr row r, as f16.
// B-fragment (column-major) for output column n is exactly Wf row n.
__global__ void k_wconv(const float* __restrict__ Wl, const float* __restrict__ Wr,
                        _Float16* __restrict__ Wf) {
  int i = blockIdx.x * blockDim.x + threadIdx.x;   // 128*128 threads
  int r = i >> 7, c = i & 127;
  Wf[r * 256 + c]       = (_Float16)Wl[i];
  Wf[r * 256 + 128 + c] = (_Float16)Wr[i];
}

// Edge-parallel scatter: 32 lanes per edge, each lane handles 4 columns.
// agg[dst] += h[src]  (sum; mean applied later via deg_inv).  agg/h are
// L2-resident (51 MB each vs 192 MB L2) so these atomics stay on-chip.
__global__ void k_scatter(const float* __restrict__ h, const int* __restrict__ src,
                          const int* __restrict__ dst, float* __restrict__ agg, int E) {
  long gid = (long)blockIdx.x * blockDim.x + threadIdx.x;
  int e = (int)(gid >> 5);
  int lane = (int)(gid & 31);
  if (e >= E) return;
  int s = src[e], d = dst[e];
  const float4 v = *(const float4*)(h + (long)s * DIM + lane * 4);
  float* ap = agg + (long)d * DIM + lane * 4;
  atomicAdd(ap + 0, v.x);
  atomicAdd(ap + 1, v.y);
  atomicAdd(ap + 2, v.z);
  atomicAdd(ap + 3, v.w);
}

// ------------------------------------------------------------- fused GEMM
// One wave computes a 16-row block of:
//   out = normalize_rows( (agg*deg_inv) @ Wl.T + bias + h @ Wr.T ) , then ReLU
// as a single K=256 fused WMMA GEMM: A' = [agg*dinv | h] (f16), B' rows = Wf.
// Weights are staged once per workgroup into LDS via async global->LDS DMA,
// stored in fragment-chunk order so every ds_load_b128 is bank-conflict-free.
__global__ __launch_bounds__(256)
void k_sage_gemm(const float* __restrict__ agg, const float* __restrict__ hin,
                 const float* __restrict__ dinv, const _Float16* __restrict__ Wf,
                 const float* __restrict__ bias, float* __restrict__ out, int n) {
  // LDS layout: 64 tiles (nt*8+kb), each tile = 64 chunks of 8 halves (16B):
  //   chunk(tile, q, lrow) at half-offset ((tile*4 + q)*16 + lrow)*8
  //   holds Wf[nt*16 + lrow][kb*32 + q*8 .. +7]
  __shared__ _Float16 sWf[64 * 64 * 8];   // 32768 halves = 64 KB

  const int tid = threadIdx.x;

  // ---- stage weights: async DMA global -> LDS, 4096 x 16B chunks
  #pragma unroll
  for (int it = 0; it < 16; ++it) {
    int cid  = it * 256 + tid;           // chunk id 0..4095
    int t    = cid >> 6;                 // tile = nt*8 + kb
    int rem  = cid & 63;
    int q    = rem >> 4;                 // 8-half sub-block within 32-k
    int lr   = rem & 15;                 // row within tile
    int nt   = t >> 3;
    int kb   = t & 7;
    const _Float16* gp = Wf + (size_t)(nt * 16 + lr) * 256 + kb * 32 + q * 8;
    unsigned ldsoff = (unsigned)(uintptr_t)(sWf + (size_t)cid * 8);
    asm volatile("global_load_async_to_lds_b128 %0, %1, off"
                 :: "v"(ldsoff), "v"(gp) : "memory");
  }
  asm volatile("s_wait_asynccnt 0x0" ::: "memory");
  __syncthreads();

  const int lane = tid & 31;
  const int wave = tid >> 5;
  const int rb   = blockIdx.x * 8 + wave;     // row-block id
  const int base = rb * 16;
  if (base >= n) return;                      // wave-uniform: EXEC stays all-1s

  const int lrow = lane & 15;                 // A-row / B-col / C-col within tile
  const int hi   = lane >> 4;                 // half-wave select
  const int koff = hi * 8;                    // 16-bit A/B layout: lo lanes K 0-7,16-23; hi lanes 8-15,24-31

  const int   row  = base + lrow;
  const float sagg = dinv[row];

  v8f acc[8];
  #pragma unroll
  for (int t = 0; t < 8; ++t) acc[t] = (v8f){0.f,0.f,0.f,0.f,0.f,0.f,0.f,0.f};

  #pragma unroll
  for (int kb = 0; kb < 8; ++kb) {            // fused K=256 in blocks of 32
    const float* ap;
    float sc;
    if (kb < 4) { ap = agg + (long)row * DIM + kb * 32;       sc = sagg; }
    else        { ap = hin + (long)row * DIM + (kb - 4) * 32; sc = 1.0f; }

    // Build A fragment in-register (f32 -> f16 with mean scaling)
    const float4 a0 = *(const float4*)(ap + koff);
    const float4 a1 = *(const float4*)(ap + koff + 4);
    const float4 a2 = *(const float4*)(ap + koff + 16);
    const float4 a3 = *(const float4*)(ap + koff + 20);
    v16h A;
    A[0]  = (_Float16)(a0.x * sc); A[1]  = (_Float16)(a0.y * sc);
    A[2]  = (_Float16)(a0.z * sc); A[3]  = (_Float16)(a0.w * sc);
    A[4]  = (_Float16)(a1.x * sc); A[5]  = (_Float16)(a1.y * sc);
    A[6]  = (_Float16)(a1.z * sc); A[7]  = (_Float16)(a1.w * sc);
    A[8]  = (_Float16)(a2.x * sc); A[9]  = (_Float16)(a2.y * sc);
    A[10] = (_Float16)(a2.z * sc); A[11] = (_Float16)(a2.w * sc);
    A[12] = (_Float16)(a3.x * sc); A[13] = (_Float16)(a3.y * sc);
    A[14] = (_Float16)(a3.z * sc); A[15] = (_Float16)(a3.w * sc);

    #pragma unroll
    for (int nt = 0; nt < 8; ++nt) {          // 8 column tiles of 16
      const int tbase = (nt * 8 + kb) * 64 * 8;               // tile base (halves)
      const v8h b0 = *(const v8h*)(sWf + tbase + ((hi    ) * 16 + lrow) * 8);
      const v8h b1 = *(const v8h*)(sWf + tbase + ((hi + 2) * 16 + lrow) * 8);
      v16h B;
      #pragma unroll
      for (int i = 0; i < 8; ++i) { B[i] = b0[i]; B[8 + i] = b1[i]; }
      acc[nt] = __builtin_amdgcn_wmma_f32_16x16x32_f16(
          /*neg_a=*/false, A, /*neg_b=*/false, B,
          /*c_mod=*/(short)0, acc[nt], /*reuse_a=*/false, /*reuse_b=*/false);
    }
  }

  // Epilogue: bias, per-row L2 norm, ReLU, store.
  #pragma unroll
  for (int nt = 0; nt < 8; ++nt) {
    const float bv = bias[nt * 16 + lrow];
    #pragma unroll
    for (int r = 0; r < 8; ++r) acc[nt][r] += bv;
  }

  const int rowbase = base + hi * 8;          // C layout: lanes 0-15 -> M=r, 16-31 -> M=r+8
  #pragma unroll
  for (int r = 0; r < 8; ++r) {
    float ss = 0.f;
    #pragma unroll
    for (int nt = 0; nt < 8; ++nt) { const float c = acc[nt][r]; ss += c * c; }
    // reduce across the 16 lanes of this half-wave (covers all 128 columns)
    ss += __shfl_xor(ss, 1, 32);
    ss += __shfl_xor(ss, 2, 32);
    ss += __shfl_xor(ss, 4, 32);
    ss += __shfl_xor(ss, 8, 32);
    const float scale = 1.0f / fmaxf(sqrtf(ss), 1e-12f);
    const long orow = rowbase + r;
    #pragma unroll
    for (int nt = 0; nt < 8; ++nt) {
      const float v = acc[nt][r] * scale;
      out[orow * DIM + nt * 16 + lrow] = fmaxf(v, 0.f);
    }
  }
}

// ------------------------------------------------------------------ launch

extern "C" void kernel_launch(void* const* d_in, const int* in_sizes, int n_in,
                              void* d_out, int out_size, void* d_ws, size_t ws_size,
                              hipStream_t stream) {
  const float* x   = (const float*)d_in[0];
  const int*   ei  = (const int*)d_in[1];
  const float* W1l = (const float*)d_in[2];
  const float* b1  = (const float*)d_in[3];
  const float* W1r = (const float*)d_in[4];
  const float* W2l = (const float*)d_in[5];
  const float* b2  = (const float*)d_in[6];
  const float* W2r = (const float*)d_in[7];
  float* out = (float*)d_out;

  const int E = in_sizes[1] / 2;
  const int n = in_sizes[0] / DIM;
  const int* src = ei;
  const int* dst = ei + E;

  // workspace carve-up (256B aligned)
  char* ws = (char*)d_ws;
  size_t off = 0;
  auto carve = [&](size_t bytes) -> void* {
    void* p = ws + off;
    off += (bytes + 255) & ~(size_t)255;
    return p;
  };
  float*    deg = (float*)carve((size_t)n * 4);
  float*    agg = (float*)carve((size_t)n * DIM * 4);
  float*    h1  = (float*)carve((size_t)n * DIM * 4);
  _Float16* Wf1 = (_Float16*)carve(128 * 256 * 2);
  _Float16* Wf2 = (_Float16*)carve(128 * 256 * 2);

  const long n4     = (long)n * DIM / 4;
  const int  zblk   = (int)((n4 + 255) / 256);
  const long d4     = (n + 3) / 4;
  const int  gemmBl = (n / 16 + 7) / 8;
  const long sthr   = (long)E * 32;
  const int  sblk   = (int)((sthr + 255) / 256);

  // degree counts (shared by both layers)
  k_zero_f4<<<(int)((d4 + 255) / 256), 256, 0, stream>>>((float4*)deg, d4);
  k_count<<<(E + 255) / 256, 256, 0, stream>>>(dst, deg, E);
  k_inv<<<(n + 255) / 256, 256, 0, stream>>>(deg, n);

  // weights -> fused f16 layout
  k_wconv<<<64, 256, 0, stream>>>(W1l, W1r, Wf1);
  k_wconv<<<64, 256, 0, stream>>>(W2l, W2r, Wf2);

  // ---- layer 1
  k_zero_f4<<<zblk, 256, 0, stream>>>((float4*)agg, n4);
  k_scatter<<<sblk, 256, 0, stream>>>(x, src, dst, agg, E);
  k_sage_gemm<<<gemmBl, 256, 0, stream>>>(agg, x, deg, Wf1, b1, h1, n);

  // ---- layer 2
  k_zero_f4<<<zblk, 256, 0, stream>>>((float4*)agg, n4);
  k_scatter<<<sblk, 256, 0, stream>>>(h1, src, dst, agg, E);
  k_sage_gemm<<<gemmBl, 256, 0, stream>>>(agg, h1, deg, Wf2, b2, out, n);
}
